// Attention_26542897889856
// MI455X (gfx1250) — matrix operand order repaired and checked by reading, not compiled, not run on hardware
//
#include <hip/hip_runtime.h>
#include <hip/hip_bf16.h>

typedef __attribute__((ext_vector_type(16))) _Float16 v16h;
typedef __attribute__((ext_vector_type(8)))  float    v8f;

#define WMMA_F16(a, b, c) \
  __builtin_amdgcn_wmma_f32_16x16x32_f16(false, (a), false, (b), (short)0, (c), false, false)

// ---------------------------------------------------------------------------
// CDNA5 async global->LDS copy (16B per lane), tracked on ASYNCcnt.
//   global_load_async_to_lds_b128 vdst(LDS byte addr), vaddr(64b global), off
// Generic LDS pointers carry the LDS byte offset in their low 32 bits.
// ---------------------------------------------------------------------------
__device__ __forceinline__ void async_copy16(const void* gptr, void* lptr) {
  unsigned lds = (unsigned)(uintptr_t)lptr;
  asm volatile("global_load_async_to_lds_b128 %0, %1, off"
               :: "v"(lds), "v"(gptr) : "memory");
}
__device__ __forceinline__ void async_wait0() {
  asm volatile("s_wait_asynccnt 0x0" ::: "memory");
}

// ---------------------------------------------------------------------------
// Fragment helpers (CDNA5 ISA 7.12.2, wave32). All reads contiguous per lane
// so they lower to ds_load_b128 / global_load_b128.
//   A 16x32 : lane m=lane&15, half=lane>>4; slot e -> k=half*16+e.
//             One 32B run.
//   B 32x16 : lane n=lane&15, kh=lane>>4; slot e -> k=kh*16+e, from a
//             transposed tile T[n][k] -> one 32B run.
//   C/D     : m = 8*(lane>>4)+r, n = lane&15
// ---------------------------------------------------------------------------
__device__ __forceinline__ v16h load_a_frag(const _Float16* tile, int ld, int lane) {
  int m = lane & 15, half = lane >> 4;
  const _Float16* p = tile + m * ld + (half << 4);
  v16h a;
#pragma unroll
  for (int e = 0; e < 16; ++e) a[e] = p[e];
  return a;
}

__device__ __forceinline__ v16h load_bt_frag(const _Float16* tileT, int ld, int lane) {
  int n = lane & 15, kh = lane >> 4;
  const _Float16* p = tileT + n * ld + (kh << 4);
  v16h b;
#pragma unroll
  for (int e = 0; e < 16; ++e) b[e] = p[e];
  return b;
}

// ---------------------------------------------------------------------------
// K0a: f32 -> f16 convert
// ---------------------------------------------------------------------------
__global__ void f32_to_f16_kernel(const float* __restrict__ src, _Float16* __restrict__ dst, int n) {
  for (int i = blockIdx.x * blockDim.x + threadIdx.x; i < n; i += gridDim.x * blockDim.x)
    dst[i] = (_Float16)src[i];
}

// K0b: f32 [R][C] -> f16 transposed [C][R]
__global__ void f32_to_f16_T_kernel(const float* __restrict__ src, _Float16* __restrict__ dst,
                                    int R, int C) {
  int total = R * C;
  for (int i = blockIdx.x * blockDim.x + threadIdx.x; i < total; i += gridDim.x * blockDim.x) {
    int r = i / C, c = i - r * C;
    dst[c * R + r] = (_Float16)src[i];
  }
}

// ---------------------------------------------------------------------------
// K1: table_h[w][h][i][j] = sum_e table[i][j][e] * W_w[e][h]
//     zero-padded to [208][224] per (w,h) so WMMA A-frags load unguarded.
// ---------------------------------------------------------------------------
__global__ void tableh_kernel(const float* __restrict__ table,
                              const float* __restrict__ wq,
                              const float* __restrict__ wk,
                              const float* __restrict__ wv,
                              _Float16* __restrict__ th) {
  const int total = 3 * 12 * 208 * 224;
  for (int idx = blockIdx.x * blockDim.x + threadIdx.x; idx < total;
       idx += gridDim.x * blockDim.x) {
    int j = idx % 224;
    int i = (idx / 224) % 208;
    int h = (idx / (224 * 208)) % 12;
    int w = idx / (224 * 208 * 12);
    float s = 0.f;
    if (i < 196 && j < 196) {
      const float* W = (w == 0) ? wq : (w == 1) ? wk : wv;
      const float* t = table + (i * 196 + j) * 12;
#pragma unroll
      for (int e = 0; e < 12; ++e) s += t[e] * W[e * 12 + h];
    }
    th[idx] = (_Float16)s;
  }
}

// ---------------------------------------------------------------------------
// K2: qkv = x @ qkv_w + b, scatter into f16 [3][B=64][H=12][N=197][64]
//     M=12608 (X padded to 12672 rows so tile loads are unguarded),
//     K=768, N=2304. Wt is transposed weights [2304][768].
//     Tiles staged via async global->LDS b128 (ASYNCcnt).
// ---------------------------------------------------------------------------
__global__ __launch_bounds__(256) void qkv_gemm_kernel(const _Float16* __restrict__ X,
                                                       const _Float16* __restrict__ Wt,
                                                       const float* __restrict__ bias,
                                                       _Float16* __restrict__ qkv) {
  __shared__ _Float16 sA[128 * 32];   // row-major [m][k]
  __shared__ _Float16 sBt[64 * 32];   // transposed [n][k]
  const int M = 12608;
  int tn = blockIdx.x, tm = blockIdx.y;
  int lane = threadIdx.x & 31, wave = threadIdx.x >> 5;
  v8f acc[4] = {};
  for (int k0 = 0; k0 < 768; k0 += 32) {
    // A tile: 128x32 halfs = 512 x 16B chunks, 2 per thread (async)
#pragma unroll
    for (int i = 0; i < 2; ++i) {
      int u = threadIdx.x + i * 256;
      int r = u >> 2, c = (u & 3) << 3;
      async_copy16(X + (size_t)(tm * 128 + r) * 768 + k0 + c, sA + r * 32 + c);
    }
    {  // B^T tile: 64x32 halfs = 256 x 16B chunks, 1 per thread (async)
      int u = threadIdx.x;
      int n = u >> 2, c = (u & 3) << 3;
      async_copy16(Wt + (size_t)(tn * 64 + n) * 768 + k0 + c, sBt + n * 32 + c);
    }
    if (k0 + 32 < 768) {  // prefetch next k-step
      __builtin_prefetch(X + (tm * 128 + (threadIdx.x & 127)) * 768 + k0 + 32, 0, 1);
      __builtin_prefetch(Wt + (tn * 64 + (threadIdx.x & 63)) * 768 + k0 + 32, 0, 1);
    }
    async_wait0();
    __syncthreads();
    v16h a = load_a_frag(sA + wave * 16 * 32, 32, lane);
#pragma unroll
    for (int j = 0; j < 4; ++j) {
      v16h b = load_bt_frag(sBt + j * 16 * 32, 32, lane);
      acc[j] = WMMA_F16(a, b, acc[j]);
    }
    __syncthreads();
  }
  int nl = lane & 15, hh = lane >> 4;
#pragma unroll
  for (int j = 0; j < 4; ++j) {
    int col = tn * 64 + j * 16 + nl;
    int t = col / 768, rem = col - t * 768;
    int h = rem >> 6, d = rem & 63;
#pragma unroll
    for (int r = 0; r < 8; ++r) {
      int row = tm * 128 + wave * 16 + hh * 8 + r;
      if (row < M) {
        int b_ = row / 197, ntok = row - b_ * 197;
        float v = acc[j][r] + bias[col];
        qkv[((((t * 64 + b_) * 12 + h) * 197 + ntok) << 6) + d] = (_Float16)v;
      }
    }
  }
}

// ---------------------------------------------------------------------------
// K3: SAVE aggregation, in place: sp = sp + th[t,h] @ sp  (per t,b,h)
//     sp staged TRANSPOSED in LDS (sSpT[64][224], k-pad zeroed) so both the
//     B-fragments and the residual read are contiguous.
// ---------------------------------------------------------------------------
__global__ __launch_bounds__(256) void agg_kernel(const _Float16* __restrict__ th,
                                                  _Float16* __restrict__ qkv) {
  __shared__ _Float16 sSpT[64 * 224];  // [d][token], tokens 196..223 = 0
  int blk = blockIdx.x;
  int h = blk % 12, b = (blk / 12) % 64, t = blk / (12 * 64);
  _Float16* sp = qkv + ((((t * 64 + b) * 12 + h) * 197) << 6) + 64;  // skip cls
  // scatter-stage transpose: coalesced b128 reads, u16 LDS scatter
  for (int u = threadIdx.x; u < 196 * 8; u += 256) {
    int r = u >> 3, seg = u & 7;
    const _Float16* g = sp + (r << 6) + (seg << 3);
    _Float16 tmp[8];
#pragma unroll
    for (int i = 0; i < 8; ++i) tmp[i] = g[i];
#pragma unroll
    for (int i = 0; i < 8; ++i) sSpT[((seg << 3) + i) * 224 + r] = tmp[i];
  }
  for (int u = threadIdx.x; u < 64 * 28; u += 256) {  // zero k-pad 196..223
    int c = u & 63, rr = 196 + (u >> 6);
    sSpT[c * 224 + rr] = (_Float16)0.f;
  }
  __syncthreads();
  const _Float16* A = th + (t * 12 + h) * 208 * 224;  // padded, unguarded loads
  int lane = threadIdx.x & 31, wave = threadIdx.x >> 5;
  int nl = lane & 15, hh = lane >> 4;
  for (int mt = wave; mt < 13; mt += 8) {
    v8f acc[4] = {};
    for (int s = 0; s < 7; ++s) {
      int k0 = s * 32;
      v16h af = load_a_frag(A + mt * 16 * 224 + k0, 224, lane);
#pragma unroll
      for (int j = 0; j < 4; ++j) {
        v16h bf = load_bt_frag(sSpT + j * 16 * 224 + k0, 224, lane);
        acc[j] = WMMA_F16(af, bf, acc[j]);
      }
    }
#pragma unroll
    for (int j = 0; j < 4; ++j) {
      int d = j * 16 + nl;
#pragma unroll
      for (int r = 0; r < 8; ++r) {
        int mg = mt * 16 + hh * 8 + r;
        if (mg < 196) {
          float v = acc[j][r] + (float)sSpT[d * 224 + mg];
          sp[(mg << 6) + d] = (_Float16)v;
        }
      }
    }
  }
}

// ---------------------------------------------------------------------------
// K4: attention per (b,h). sK row-major (S B-frags contiguous by K rows),
//     sV transposed (P@V B-frags contiguous), per-wave P patch as f16
//     A-fragments. f32 softmax with shfl_xor over 16-lane halves.
// ---------------------------------------------------------------------------
__global__ __launch_bounds__(256) void attn_kernel(const _Float16* __restrict__ qkv,
                                                   _Float16* __restrict__ ctx) {
  __shared__ _Float16 sK[208 * 64];       // [token][d], tokens 197..207 = 0
  __shared__ _Float16 sVt[64 * 224];      // [d][token], tokens 197..223 = 0
  __shared__ _Float16 sP[8 * 16 * 224];   // per-wave P patch
  int b = blockIdx.x / 12, h = blockIdx.x % 12;
  const _Float16* Q  = qkv + (((((0 * 64) + b) * 12 + h) * 197) << 6);
  const _Float16* Km = qkv + (((((1 * 64) + b) * 12 + h) * 197) << 6);
  const _Float16* Vm = qkv + (((((2 * 64) + b) * 12 + h) * 197) << 6);
  for (int u = threadIdx.x; u < 208 * 8; u += 256) {  // K rows, b128 copies
    int r = u >> 3, seg = u & 7;
    unsigned int w0 = 0, w1 = 0, w2 = 0, w3 = 0;
    if (r < 197) {
      const unsigned int* g = (const unsigned int*)(Km + (r << 6) + (seg << 3));
      w0 = g[0]; w1 = g[1]; w2 = g[2]; w3 = g[3];
    }
    unsigned int* d = (unsigned int*)(sK + (r << 6) + (seg << 3));
    d[0] = w0; d[1] = w1; d[2] = w2; d[3] = w3;
  }
  for (int u = threadIdx.x; u < 197 * 8; u += 256) {  // V transpose-scatter
    int r = u >> 3, seg = u & 7;
    const _Float16* g = Vm + (r << 6) + (seg << 3);
    _Float16 tmp[8];
#pragma unroll
    for (int i = 0; i < 8; ++i) tmp[i] = g[i];
#pragma unroll
    for (int i = 0; i < 8; ++i) sVt[((seg << 3) + i) * 224 + r] = tmp[i];
  }
  for (int u = threadIdx.x; u < 64 * 27; u += 256) {  // zero V k-pad 197..223
    int c = u % 64, rr = 197 + u / 64;
    sVt[c * 224 + rr] = (_Float16)0.f;
  }
  __syncthreads();
  int lane = threadIdx.x & 31, wave = threadIdx.x >> 5;
  int nl = lane & 15, hh = lane >> 4;
  _Float16* pw = sP + wave * 16 * 224;
  const float scale = 0.125f;  // 64^-0.5

  for (int qt = wave; qt < 13; qt += 8) {
    // Q fragments: unguarded contiguous b128 (over-read stays in workspace,
    // garbage rows masked at the store)
    v16h aq[2];
#pragma unroll
    for (int s = 0; s < 2; ++s)
      aq[s] = load_a_frag(Q + qt * 16 * 64 + s * 32, 64, lane);

    // S = Q @ K^T over 13 column tiles
    v8f sacc[13] = {};
#pragma unroll
    for (int jt = 0; jt < 13; ++jt)
#pragma unroll
      for (int s = 0; s < 2; ++s) {
        // B-frag: lane n reads K row (jt*16+n), contiguous 16 halfs
        v16h bf;
        const _Float16* p = sK + ((jt * 16 + nl) << 6) + s * 32 + (hh << 4);
#pragma unroll
        for (int e = 0; e < 16; ++e) bf[e] = p[e];
        sacc[jt] = WMMA_F16(aq[s], bf, sacc[jt]);
      }

    // softmax over 208 cols (197 valid); rows live in 16-lane halves
#pragma unroll
    for (int r = 0; r < 8; ++r) {
      float mx = -1e30f;
#pragma unroll
      for (int jt = 0; jt < 13; ++jt) {
        int col = jt * 16 + nl;
        float s = (col < 197) ? sacc[jt][r] : -1e30f;
        mx = fmaxf(mx, s);
      }
#pragma unroll
      for (int off = 1; off < 16; off <<= 1) mx = fmaxf(mx, __shfl_xor(mx, off, 32));
      mx *= scale;
      float sum = 0.f;
#pragma unroll
      for (int jt = 0; jt < 13; ++jt) {
        int col = jt * 16 + nl;
        float p = (col < 197) ? __expf(sacc[jt][r] * scale - mx) : 0.f;
        sacc[jt][r] = p;
        sum += p;
      }
#pragma unroll
      for (int off = 1; off < 16; off <<= 1) sum += __shfl_xor(sum, off, 32);
      float inv = 1.0f / sum;
#pragma unroll
      for (int jt = 0; jt < 13; ++jt) sacc[jt][r] *= inv;
#pragma unroll
      for (int jt = 0; jt < 13; ++jt)
        pw[(hh * 8 + r) * 224 + jt * 16 + nl] = (_Float16)sacc[jt][r];
    }
    {  // zero pad P cols 208..223
      int row = lane & 15, c0 = 208 + (hh << 3);
#pragma unroll
      for (int c = 0; c < 8; ++c) pw[row * 224 + c0 + c] = (_Float16)0.f;
    }

    // out = P @ V : 16x64 from 16x224 @ 224x64 (V transposed in LDS)
    v8f oacc[4] = {};
    for (int s = 0; s < 7; ++s) {
      int k0 = s * 32;
      v16h ap = load_a_frag(pw + k0, 224, lane);
#pragma unroll
      for (int j = 0; j < 4; ++j) {
        v16h bf = load_bt_frag(sVt + j * 16 * 224 + k0, 224, lane);
        oacc[j] = WMMA_F16(ap, bf, oacc[j]);
      }
    }
#pragma unroll
    for (int j = 0; j < 4; ++j) {
      int d = j * 16 + nl;
#pragma unroll
      for (int r = 0; r < 8; ++r) {
        int mg = qt * 16 + hh * 8 + r;
        if (mg < 197)
          ctx[(b * 197 + mg) * 768 + (h << 6) + d] = (_Float16)oacc[j][r];
      }
    }
  }
}

// ---------------------------------------------------------------------------
// K5: out = ctx @ proj_w + proj_b (f32 out). M=12608 (ctx padded to 12672),
//     K=768, N=768. Async-staged tiles like K2.
// ---------------------------------------------------------------------------
__global__ __launch_bounds__(256) void proj_gemm_kernel(const _Float16* __restrict__ X,
                                                        const _Float16* __restrict__ Wt,
                                                        const float* __restrict__ bias,
                                                        float* __restrict__ out) {
  __shared__ _Float16 sA[128 * 32];
  __shared__ _Float16 sBt[64 * 32];
  const int M = 12608;
  int tn = blockIdx.x, tm = blockIdx.y;
  int lane = threadIdx.x & 31, wave = threadIdx.x >> 5;
  v8f acc[4] = {};
  for (int k0 = 0; k0 < 768; k0 += 32) {
#pragma unroll
    for (int i = 0; i < 2; ++i) {
      int u = threadIdx.x + i * 256;
      int r = u >> 2, c = (u & 3) << 3;
      async_copy16(X + (size_t)(tm * 128 + r) * 768 + k0 + c, sA + r * 32 + c);
    }
    {
      int u = threadIdx.x;
      int n = u >> 2, c = (u & 3) << 3;
      async_copy16(Wt + (size_t)(tn * 64 + n) * 768 + k0 + c, sBt + n * 32 + c);
    }
    if (k0 + 32 < 768) {
      __builtin_prefetch(X + (tm * 128 + (threadIdx.x & 127)) * 768 + k0 + 32, 0, 1);
      __builtin_prefetch(Wt + (tn * 64 + (threadIdx.x & 63)) * 768 + k0 + 32, 0, 1);
    }
    async_wait0();
    __syncthreads();
    v16h a = load_a_frag(sA + wave * 16 * 32, 32, lane);
#pragma unroll
    for (int j = 0; j < 4; ++j) {
      v16h b = load_bt_frag(sBt + j * 16 * 32, 32, lane);
      acc[j] = WMMA_F16(a, b, acc[j]);
    }
    __syncthreads();
  }
  int nl = lane & 15, hh = lane >> 4;
#pragma unroll
  for (int j = 0; j < 4; ++j) {
    int col = tn * 64 + j * 16 + nl;
#pragma unroll
    for (int r = 0; r < 8; ++r) {
      int row = tm * 128 + wave * 16 + hh * 8 + r;
      if (row < M) out[row * 768 + col] = acc[j][r] + bias[col];
    }
  }
}

// ---------------------------------------------------------------------------
// Host launcher
// ---------------------------------------------------------------------------
extern "C" void kernel_launch(void* const* d_in, const int* in_sizes, int n_in,
                              void* d_out, int out_size, void* d_ws, size_t ws_size,
                              hipStream_t stream) {
  (void)in_sizes; (void)n_in; (void)out_size; (void)ws_size;
  const float* x      = (const float*)d_in[0];
  const float* qkv_w  = (const float*)d_in[1];
  const float* qkv_b  = (const float*)d_in[2];
  const float* proj_w = (const float*)d_in[3];
  const float* proj_b = (const float*)d_in[4];
  const float* wq     = (const float*)d_in[5];
  const float* wk     = (const float*)d_in[6];
  const float* wv     = (const float*)d_in[7];
  const float* table  = (const float*)d_in[8];
  float* out = (float*)d_out;

  char* ws = (char*)d_ws;
  size_t off = 0;
  auto walloc = [&](size_t bytes) {
    void* p = ws + off;
    off = (off + bytes + 255) & ~(size_t)255;
    return p;
  };
  const size_t MN  = 12608;   // 64*197 valid rows
  const size_t MNP = 12672;   // padded to 99*128 so GEMM tiles load unguarded
  _Float16* xh     = (_Float16*)walloc(MNP * 768 * 2);
  _Float16* qkvwt  = (_Float16*)walloc(2304UL * 768 * 2);           // transposed
  _Float16* projwt = (_Float16*)walloc(768UL * 768 * 2);            // transposed
  _Float16* qkvh   = (_Float16*)walloc(3UL * 64 * 12 * 197 * 64 * 2);
  _Float16* ctxh   = (_Float16*)walloc(MNP * 768 * 2);
  _Float16* thp    = (_Float16*)walloc(3UL * 12 * 208 * 224 * 2);   // padded

  // K0: converts (+ weight transposes)
  f32_to_f16_kernel<<<4096, 256, 0, stream>>>(x, xh, (int)(MN * 768));
  f32_to_f16_T_kernel<<<2048, 256, 0, stream>>>(qkv_w, qkvwt, 768, 2304);
  f32_to_f16_T_kernel<<<1024, 256, 0, stream>>>(proj_w, projwt, 768, 768);

  // K1: padded table_h
  tableh_kernel<<<2048, 256, 0, stream>>>(table, wq, wk, wv, thp);

  // K2: qkv GEMM (36 x 99 tiles)
  qkv_gemm_kernel<<<dim3(36, 99), 256, 0, stream>>>(xh, qkvwt, qkv_b, qkvh);

  // K3: SAVE aggregation (3*64*12 blocks)
  agg_kernel<<<3 * 64 * 12, 256, 0, stream>>>(thp, qkvh);

  // K4: attention (64*12 blocks)
  attn_kernel<<<64 * 12, 256, 0, stream>>>(qkvh, ctxh);

  // K5: output projection (12 x 99 tiles)
  proj_gemm_kernel<<<dim3(12, 99), 256, 0, stream>>>(ctxh, projwt, proj_b, out);
}
